// RNN_73177652789691
// MI455X (gfx1250) — matrix operand order repaired and checked by reading, not compiled, not run on hardware
//
#include <hip/hip_runtime.h>
#include <hip/hip_bf16.h>

typedef __attribute__((ext_vector_type(16))) __bf16 v16bf;
typedef __attribute__((ext_vector_type(8)))  __bf16 v8bf;
typedef __attribute__((ext_vector_type(8)))  float  v8f;

#define T_STEPS 2048
#define HDIM    256
#define BATCH   32
#define LDSS    264   // padded row stride (bf16 elems): 528B -> lanes 4 banks apart

// ---------------- helpers ----------------

__device__ __forceinline__ float fast_tanh(float x) {
    // tanh(x) = (e^{2x}-1)/(e^{2x}+1), e^{2x} = exp2(2*log2(e)*x)
    float xc = fminf(fmaxf(x, -15.0f), 15.0f);
    float e  = __builtin_amdgcn_exp2f(xc * 2.8853900817779268f);
    return (e - 1.0f) * __builtin_amdgcn_rcpf(e + 1.0f);
}

__device__ __forceinline__ v8f wmma_acc(v16bf a, v16bf b, v8f c) {
    // D = A(16x32 bf16) * B(32x16 bf16) + C(16x16 f32)
    return __builtin_amdgcn_wmma_f32_16x16x32_bf16(
        /*neg_a=*/false, a, /*neg_b=*/false, b,
        /*c_mod=*/(short)0, c, /*reuse_a=*/false, /*reuse_b=*/false);
}

// A-fragment (16x32 bf16) from LDS, row-major h buffer with stride LDSS.
// ISA layout: lane(0-15) row=lane, k = half*8 + {0..7} (v0-3) and 16+half*8+{0..7} (v4-7)
__device__ __forceinline__ v16bf ldsA(const __bf16* h, int row, int kElem, int half) {
    const v8bf* lo = (const v8bf*)(h + row * LDSS + kElem + half * 8);
    const v8bf* hi = (const v8bf*)(h + row * LDSS + kElem + 16 + half * 8);
    v8bf L = *lo, H = *hi;
    v16bf a;
#pragma unroll
    for (int j = 0; j < 8; ++j) { a[j] = L[j]; a[8 + j] = H[j]; }
    return a;
}

// A-fragment built from f32 global x (row-major, row stride 256)
__device__ __forceinline__ v16bf ldx_frag(const float* __restrict__ x, int brow, int t, int kbase) {
    const float4* p = (const float4*)(x + ((size_t)brow * T_STEPS + t) * HDIM + kbase);
    float4 f0 = p[0], f1 = p[1], f2 = p[4], f3 = p[5];
    float tmp[16] = { f0.x, f0.y, f0.z, f0.w, f1.x, f1.y, f1.z, f1.w,
                      f2.x, f2.y, f2.z, f2.w, f3.x, f3.y, f3.z, f3.w };
    v16bf a;
#pragma unroll
    for (int j = 0; j < 16; ++j) a[j] = (__bf16)tmp[j];
    return a;
}

// ---------------- kernel 1: pack weights into per-lane B-fragment layout ----------------
// B-frag (32x16 bf16, KxN): lanes 0-15: N=lane, K=0..7,16..23; lanes 16-31: N=lane-16, K=8..15,24..31.
// Packed slot for (strip s, ktile kt, lane): 16 bf16 = W[kt*32 + (lane>=16)*8 + j + (j&8)][s*16 + lane%16]
__global__ void pack_w(const float* __restrict__ W0, const float* __restrict__ W1,
                       __bf16* __restrict__ pw0in, __bf16* __restrict__ pw0rec,
                       __bf16* __restrict__ pw1) {
    int gid = blockIdx.x * blockDim.x + threadIdx.x;   // 0..16383
    if (gid >= 512 * 32) return;
    int lane = gid & 31;
    int f    = gid >> 5;                                // fragment id 0..511
    const float* W; int krow0, scol, slot; v16bf* dst;
    if (f < 128)       { W = W0; krow0 =        (f & 7) * 32;  scol = (f >> 3) * 16;  slot = f;        dst = (v16bf*)pw0in;  }
    else if (f < 256)  { int fl = f - 128; W = W0; krow0 = 256 + (fl & 7) * 32; scol = (fl >> 3) * 16; slot = fl; dst = (v16bf*)pw0rec; }
    else               { int fl = f - 256; W = W1; krow0 =       (fl & 15) * 32; scol = (fl >> 4) * 16; slot = fl; dst = (v16bf*)pw1;  }
    int n  = scol + (lane & 15);
    int kb = krow0 + ((lane >> 4) << 3);
    v16bf v;
#pragma unroll
    for (int j = 0; j < 16; ++j) v[j] = (__bf16)W[(kb + j + (j & 8)) * HDIM + n];
    dst[slot * 32 + lane] = v;
}

// ---------------- kernel 2: bulk input projection pre0 = x @ W0[:256] + b0 ----------------
// grid = T, block = 512 (16 waves); wave s owns cols [16s,16s+16), 2 M-tiles (batch 0-15, 16-31)
__global__ __launch_bounds__(512) void pregemm(const float* __restrict__ x,
                                               const float* __restrict__ b0,
                                               const __bf16* __restrict__ pw0in,
                                               float* __restrict__ pre0) {
    int t = blockIdx.x;
    int tid = threadIdx.x, lane = tid & 31, s = tid >> 5, half = lane >> 4, lq = lane & 15;
    int col = s * 16 + lq;
    float bias = b0[col];
    v8f c0, c1;
#pragma unroll
    for (int i = 0; i < 8; ++i) { c0[i] = bias; c1[i] = bias; }
    const v16bf* wp = (const v16bf*)pw0in;
#pragma unroll
    for (int kt = 0; kt < 8; ++kt) {
        v16bf w  = wp[(s * 8 + kt) * 32 + lane];
        v16bf a0 = ldx_frag(x, lq,      t, kt * 32 + half * 8);
        v16bf a1 = ldx_frag(x, 16 + lq, t, kt * 32 + half * 8);
        c0 = wmma_acc(a0, w, c0);
        c1 = wmma_acc(a1, w, c1);
    }
    // C layout: vgpr i -> m = i + 8*half, n = col
#pragma unroll
    for (int i = 0; i < 8; ++i) {
        int m = i + 8 * half;
        pre0[((size_t)t * BATCH + m)      * HDIM + col] = c0[i];
        pre0[((size_t)t * BATCH + 16 + m) * HDIM + col] = c1[i];
    }
}

// ---------------- kernel 3: serial recurrence, 2 independent WGs x 16 batch rows ----------------
// Double-buffered h state (parity flips each step) -> only 2 barriers per timestep.
// pre0[t+1] prefetched into registers one full iteration ahead.
__global__ __launch_bounds__(512) void rnn_seq(const float* __restrict__ h0_in,
                                               const float* __restrict__ b1,
                                               const __bf16* __restrict__ pw0rec,
                                               const __bf16* __restrict__ pw1,
                                               const float* __restrict__ pre0,
                                               float* __restrict__ out_hs,
                                               float* __restrict__ out_ht) {
    __shared__ __align__(16) __bf16 h0s[2][16 * LDSS];
    __shared__ __align__(16) __bf16 h1s[2][16 * LDSS];
    int bIdx = blockIdx.x * 16;
    int tid = threadIdx.x, lane = tid & 31, s = tid >> 5, half = lane >> 4, lq = lane & 15;
    int col = s * 16 + lq;
    int mrow = 8 * half;   // C vgpr i -> local batch row i + mrow

    // init h state in LDS (bf16), parity 0
    for (int idx = tid; idx < 16 * 256; idx += 512) {
        int b = idx >> 8, k = idx & 255;
        h0s[0][b * LDSS + k] = (__bf16)h0_in[(size_t)(bIdx + b) * HDIM + k];          // layer 0
        h1s[0][b * LDSS + k] = (__bf16)h0_in[(size_t)(BATCH + bIdx + b) * HDIM + k];  // layer 1
    }
    float bias1 = b1[col];
    const v16bf* w0p = (const v16bf*)pw0rec;
    const v16bf* w1p = (const v16bf*)pw1;

    // preload pre0[t=0] into registers
    float cn[8];
#pragma unroll
    for (int i = 0; i < 8; ++i)
        cn[i] = pre0[((size_t)0 * BATCH + bIdx + i + mrow) * HDIM + col];

    __syncthreads();

    float h1v[8];
    int p = 0;
    for (int t = 0; t < T_STEPS; ++t) {
        // ---- layer 0: z0 = pre0[t] + h0 @ W0rec ----
        v8f c0;
#pragma unroll
        for (int i = 0; i < 8; ++i) c0[i] = cn[i];

        // prefetch pre0[t+1] (clamped on last iter; value unused then)
        int tn = (t + 1 < T_STEPS) ? (t + 1) : t;
#pragma unroll
        for (int i = 0; i < 8; ++i)
            cn[i] = pre0[((size_t)tn * BATCH + bIdx + i + mrow) * HDIM + col];

#pragma unroll
        for (int kt = 0; kt < 8; ++kt) {
            v16bf a = ldsA(h0s[p], lq, kt * 32, half);
            v16bf w = w0p[(s * 8 + kt) * 32 + lane];
            c0 = wmma_acc(a, w, c0);
        }
        float h0v[8];
#pragma unroll
        for (int i = 0; i < 8; ++i) h0v[i] = fast_tanh(c0[i]);

        // publish h0_new into the OTHER buffer (nobody reads it yet)
#pragma unroll
        for (int i = 0; i < 8; ++i) h0s[1 - p][(i + mrow) * LDSS + col] = (__bf16)h0v[i];
        __syncthreads();                       // BAR1: h0[1-p] visible to all

        // ---- layer 1: z1 = [h0_new | h1] @ W1 + b1 ----
        v8f c1;
#pragma unroll
        for (int i = 0; i < 8; ++i) c1[i] = bias1;
#pragma unroll
        for (int kt = 0; kt < 16; ++kt) {
            v16bf a = (kt < 8) ? ldsA(h0s[1 - p], lq, kt * 32, half)
                               : ldsA(h1s[p], lq, (kt - 8) * 32, half);
            v16bf w = w1p[(s * 16 + kt) * 32 + lane];
            c1 = wmma_acc(a, w, c1);
        }
#pragma unroll
        for (int i = 0; i < 8; ++i) h1v[i] = fast_tanh(c1[i]);

        // hidden_states[b, t, col] (B,T,H)
#pragma unroll
        for (int i = 0; i < 8; ++i)
            out_hs[((size_t)(bIdx + i + mrow) * T_STEPS + t) * HDIM + col] = h1v[i];

        // publish h1_new into the OTHER buffer
#pragma unroll
        for (int i = 0; i < 8; ++i) h1s[1 - p][(i + mrow) * LDSS + col] = (__bf16)h1v[i];
        __syncthreads();                       // BAR2: h1[1-p] visible before next step

        p ^= 1;
    }

    // final layer-1 state (B,1,H), exact f32 from last step's registers
#pragma unroll
    for (int i = 0; i < 8; ++i)
        out_ht[(size_t)(bIdx + i + mrow) * HDIM + col] = h1v[i];
}

// ---------------- launcher ----------------
extern "C" void kernel_launch(void* const* d_in, const int* in_sizes, int n_in,
                              void* d_out, int out_size, void* d_ws, size_t ws_size,
                              hipStream_t stream) {
    const float* x  = (const float*)d_in[0];   // (B,T,E)
    const float* h0 = (const float*)d_in[1];   // (L,B,H)
    const float* W0 = (const float*)d_in[2];   // (E+H, H)
    const float* b0 = (const float*)d_in[3];   // (H)
    const float* W1 = (const float*)d_in[4];   // (2H, H)
    const float* b1 = (const float*)d_in[5];   // (H)

    char* ws = (char*)d_ws;
    __bf16* pw0in  = (__bf16*)(ws + 0);        // 128 KB packed bf16 fragments
    __bf16* pw0rec = (__bf16*)(ws + 131072);   // 128 KB
    __bf16* pw1    = (__bf16*)(ws + 262144);   // 256 KB
    float*  pre0   = (float*)(ws + 524288);    // T*B*H f32 = 64 MB

    float* out_hs = (float*)d_out;                                   // (B,T,H)
    float* out_ht = (float*)d_out + (size_t)BATCH * T_STEPS * HDIM;  // (B,1,H)

    pack_w<<<64, 256, 0, stream>>>(W0, W1, pw0in, pw0rec, pw1);
    pregemm<<<T_STEPS, 512, 0, stream>>>(x, b0, pw0in, pre0);
    rnn_seq<<<2, 512, 0, stream>>>(h0, b1, pw0rec, pw1, pre0, out_hs, out_ht);
}
